// GraphTransformerLayer_38628935860832
// MI455X (gfx1250) — matrix operand order repaired and checked by reading, not compiled, not run on hardware
//
#include <hip/hip_runtime.h>
#include <hip/hip_bf16.h>

// ---------------------------------------------------------------------------
// GraphTransformerLayer for MI455X (gfx1250), wave32 + WMMA bf16.
//
//   wt  : f32 weights -> bf16 W^T [N,K]            (tiny)
//   cvt : x, edge_attr -> bf16                     (bandwidth)
//   gemm q/k/v, edge_proj                          (WMMA, LDS-staged B)
//   logits -> atomicMax -> exp/atomicAdd -> msgs   (scatter softmax, L2 atomics)
//   gemm out-proj  + fused (+x, LN1)      -> h
//   gemm ffn1(GELU) -> ffn2 + fused (+h, LN2) -> h2 (d_out part 1)
//   gemm eu1 (gathered concat A, GELU) -> eu2 + fused (+edge_attr, LN_e)
//                                          -> e_out (d_out part 2)
// ---------------------------------------------------------------------------

typedef __bf16 bfx;
typedef bfx   v16bf __attribute__((ext_vector_type(16)));
typedef bfx   v8bf  __attribute__((ext_vector_type(8)));
typedef bfx   v4bf  __attribute__((ext_vector_type(4)));
typedef float v8f   __attribute__((ext_vector_type(8)));

#define NNODES 20000
#define NEDGES 640000
#define DIM    128

__device__ __forceinline__ float gelu_f(float x) {
  return 0.5f * x * (1.0f + erff(x * 0.7071067811865475f));
}
// monotonic float<->uint key for atomicMax on floats
__device__ __forceinline__ unsigned fkey(float f) {
  unsigned u = __float_as_uint(f);
  return (u & 0x80000000u) ? ~u : (u | 0x80000000u);
}
__device__ __forceinline__ float funkey(unsigned k) {
  unsigned u = (k & 0x80000000u) ? (k & 0x7fffffffu) : ~k;
  return __uint_as_float(u);
}
__device__ __forceinline__ v8bf ld8(const bfx* p) {
  return *reinterpret_cast<const v8bf*>(p);
}
#define CAT16(lo, hi)                                                         \
  __builtin_shufflevector(lo, hi, 0, 1, 2, 3, 4, 5, 6, 7, 8, 9, 10, 11, 12,   \
                          13, 14, 15)

// ---------------------------------------------------------------------------
// WMMA GEMM: C[M,N] = A[M,K](bf16) * W[K,N] + bias, W held as Wt[N,K] bf16.
// Block = 128 threads = 4 waves; each wave computes a 32x128 strip (2 M-frags
// x 8 N-frags). B is staged per 128-deep K chunk into LDS in fragment order
// (1KB frags, lane*32B -> conflict-free ds_load_b128), and read back with a
// 2-fragment in-register lookahead so ds latency overlaps the WMMAs. A is
// double-buffered from global; the K loop is unrolled 2x so the two A buffers
// ping-pong (no register copies, no WMMA->VALU hazard nops). Full output rows
// are wave-resident, enabling the fused residual+LayerNorm epilogue.
// EPI: 0 = bias(+GELU) store f32/bf16;  1 = bias + residual + LayerNorm.
// GATHER: A row m = [h2b[src[m]] | h2b[dst[m]] | eab[m]]  (K = 384).
// K is always a multiple of 64 (128 / 256 / 384).
// ---------------------------------------------------------------------------
template <int EPI, bool GATHER, bool GELU>
__global__ __launch_bounds__(128) void gemm_wmma_k(
    const bfx* __restrict__ A, int lda, const bfx* __restrict__ A2,
    const long long* __restrict__ srcI, const long long* __restrict__ dstI,
    const bfx* __restrict__ Wt, const float* __restrict__ bias,
    const float* __restrict__ residual, const float* __restrict__ gamma,
    const float* __restrict__ beta, float* __restrict__ outF,
    bfx* __restrict__ outB, int M, int K, int Nout) {
  __shared__ bfx smem[16384];  // 32 KB: one 128-deep K chunk of B fragments

  const int wave  = threadIdx.x >> 5;
  const int lane  = threadIdx.x & 31;
  const int l15   = lane & 15;
  const int lhalf = lane >> 4;
  const int m0    = blockIdx.x * 128 + wave * 32;
  const int n0    = blockIdx.y * 128;

  int mr0 = m0 + l15;      if (mr0 > M - 1) mr0 = M - 1;  // clamp (stores guarded)
  int mr1 = m0 + 16 + l15; if (mr1 > M - 1) mr1 = M - 1;

  const bfx *a0s0, *a0s1 = nullptr, *a0s2 = nullptr;
  const bfx *a1s0, *a1s1 = nullptr, *a1s2 = nullptr;
  if (GATHER) {
    a0s0 = A + srcI[mr0] * 128;  a0s1 = A + dstI[mr0] * 128;
    a0s2 = A2 + (long long)mr0 * 128;
    a1s0 = A + srcI[mr1] * 128;  a1s1 = A + dstI[mr1] * 128;
    a1s2 = A2 + (long long)mr1 * 128;
  } else {
    a0s0 = A + (long long)mr0 * lda;
    a1s0 = A + (long long)mr1 * lda;
  }

  // A fragment (16x32 bf16): lanes 0-15 hold k in [0..7]+[16..23],
  // lanes 16-31 hold k in [8..15]+[24..31]  (ISA 7.12.2 layout)
  auto loadA = [&](const bfx* s0, const bfx* s1, const bfx* s2,
                   int k0) -> v16bf {
    const bfx* ap;
    int ko;
    if (GATHER) {
      ap = (k0 < 128) ? s0 : (k0 < 256) ? s1 : s2;
      ko = k0 & 127;
    } else {
      ap = s0;
      ko = k0;
    }
    const v8bf lo = ld8(ap + ko + lhalf * 8);
    const v8bf hi = ld8(ap + ko + 16 + lhalf * 8);
    return CAT16(lo, hi);
  };

  v8f acc[2][8];
#pragma unroll
  for (int mf = 0; mf < 2; ++mf)
#pragma unroll
    for (int i = 0; i < 8; ++i)
      acc[mf][i] = (v8f){0.f, 0.f, 0.f, 0.f, 0.f, 0.f, 0.f, 0.f};

  // B fragment (32x16 bf16): lanes 0-15 hold k in [0..7]+[16..23], lanes 16-31 hold k in [8..15]+[24..31]
  auto loadB = [&](const bfx* bbase, int nf) -> v16bf {
    const bfx* bp = bbase + nf * 512 + lane * 16;
    const v8bf blo = *reinterpret_cast<const v8bf*>(bp);
    const v8bf bhi = *reinterpret_cast<const v8bf*>(bp + 8);
    return CAT16(blo, bhi);
  };

  // 8 WMMA-pairs against the LDS-resident B fragments of k-step `ks`,
  // with a 2-fragment in-register lookahead (ds latency hidden by WMMAs)
  auto mma_step = [&](int ks, const v16bf& x0, const v16bf& x1) {
    const bfx* bbase = smem + ((ks >> 5) & 3) * 4096;  // 8 frags * 512 elems
    v16bf b0 = loadB(bbase, 0);
    v16bf b1 = loadB(bbase, 1);
#pragma unroll
    for (int nf = 0; nf < 8; nf += 2) {
      v16bf t0 = b0, t1 = b1;
      if (nf + 2 < 8) t0 = loadB(bbase, nf + 2);
      acc[0][nf] = __builtin_amdgcn_wmma_f32_16x16x32_bf16(
          false, x0, false, b0, (short)0, acc[0][nf], false, false);
      acc[1][nf] = __builtin_amdgcn_wmma_f32_16x16x32_bf16(
          false, x1, false, b0, (short)0, acc[1][nf], false, false);
      if (nf + 3 < 8) t1 = loadB(bbase, nf + 3);
      acc[0][nf + 1] = __builtin_amdgcn_wmma_f32_16x16x32_bf16(
          false, x0, false, b1, (short)0, acc[0][nf + 1], false, false);
      acc[1][nf + 1] = __builtin_amdgcn_wmma_f32_16x16x32_bf16(
          false, x1, false, b1, (short)0, acc[1][nf + 1], false, false);
      b0 = t0;
      b1 = t1;
    }
  };

  v16bf a0 = loadA(a0s0, a0s1, a0s2, 0);
  v16bf a1 = loadA(a1s0, a1s1, a1s2, 0);

  for (int k0 = 0; k0 < K; k0 += 64) {
    if ((k0 & 127) == 0) {
      __syncthreads();  // protect previous chunk's readers
      // stage chunk [k0, k0+128): 32 frags of 1 KB, 8 per wave, swizzled so
      // frag reads are ds_load_b128 with consecutive lanes -> no bank conflicts
#pragma unroll
      for (int i = 0; i < 8; ++i) {
        const int f  = wave + i * 4;
        const int nf = f & 7;
        const int kk = k0 + (f >> 3) * 32;
        const bfx* wp =
            Wt + (long long)(n0 + nf * 16 + l15) * K + kk + lhalf * 8;
        const v8bf w0 = ld8(wp), w1 = ld8(wp + 16);
        bfx* dp = smem + f * 512 + lane * 16;
        *reinterpret_cast<v8bf*>(dp) = w0;
        *reinterpret_cast<v8bf*>(dp + 8) = w1;
      }
      __syncthreads();
    }

    // half 1: consume a0/a1 at k0 while fetching k0+32 (always in range)
    if (!GATHER) {
      __builtin_prefetch(a0s0 + k0 + 96, 0, 1);  // global_prefetch_b8
      __builtin_prefetch(a1s0 + k0 + 96, 0, 1);
    }
    const v16bf c0 = loadA(a0s0, a0s1, a0s2, k0 + 32);
    const v16bf c1 = loadA(a1s0, a1s1, a1s2, k0 + 32);
    mma_step(k0, a0, a1);

    // half 2: consume c0/c1 at k0+32 while fetching k0+64 into a0/a1
    if (k0 + 64 < K) {
      a0 = loadA(a0s0, a0s1, a0s2, k0 + 64);
      a1 = loadA(a1s0, a1s1, a1s2, k0 + 64);
    }
    mma_step(k0 + 32, c0, c1);
  }

  if (EPI == 0) {
#pragma unroll
    for (int mf = 0; mf < 2; ++mf)
#pragma unroll
      for (int nf = 0; nf < 8; ++nf) {
        const int col = n0 + nf * 16 + l15;
        const float bv = bias[col];
#pragma unroll
        for (int r = 0; r < 8; ++r) {
          const int row = m0 + mf * 16 + r + lhalf * 8;  // C: vgpr r -> M=r(+8)
          if (row < M) {
            float vx = acc[mf][nf][r] + bv;
            if (GELU) vx = gelu_f(vx);
            if (outF) outF[(long long)row * Nout + col] = vx;
            if (outB) outB[(long long)row * Nout + col] = (bfx)vx;
          }
        }
      }
  } else {
    // fused residual + LayerNorm over the wave-resident 128-wide rows
#pragma unroll
    for (int mf = 0; mf < 2; ++mf) {
      float rs[8], rq[8];
#pragma unroll
      for (int r = 0; r < 8; ++r) { rs[r] = 0.f; rq[r] = 0.f; }
#pragma unroll
      for (int nf = 0; nf < 8; ++nf) {
        const int col = nf * 16 + l15;
        const float bv = bias[col];
#pragma unroll
        for (int r = 0; r < 8; ++r) {
          int row = m0 + mf * 16 + r + lhalf * 8;
          if (row > M - 1) row = M - 1;
          const float vx =
              acc[mf][nf][r] + bv + residual[(long long)row * 128 + col];
          acc[mf][nf][r] = vx;
          rs[r] += vx;
          rq[r] += vx * vx;
        }
      }
#pragma unroll
      for (int r = 0; r < 8; ++r) {
        float s1 = rs[r], s2 = rq[r];
#pragma unroll
        for (int off = 1; off < 16; off <<= 1) {  // reduce over the 16-lane half
          s1 += __shfl_xor(s1, off, 32);
          s2 += __shfl_xor(s2, off, 32);
        }
        const float mean = s1 * (1.0f / 128.0f);
        const float var = s2 * (1.0f / 128.0f) - mean * mean;
        rs[r] = mean;
        rq[r] = rsqrtf(var + 1e-5f);
      }
#pragma unroll
      for (int nf = 0; nf < 8; ++nf) {
        const int col = nf * 16 + l15;
        const float g = gamma[col], bb = beta[col];
#pragma unroll
        for (int r = 0; r < 8; ++r) {
          const int row = m0 + mf * 16 + r + lhalf * 8;
          if (row < M) {
            const float o = (acc[mf][nf][r] - rs[r]) * rq[r] * g + bb;
            outF[(long long)row * 128 + col] = o;
            if (outB) outB[(long long)row * 128 + col] = (bfx)o;
          }
        }
      }
    }
  }
}

// ---------------------------------------------------------------------------
// Elementwise / scatter kernels
// ---------------------------------------------------------------------------
__global__ void wt_bf16_k(const float* __restrict__ W, bfx* __restrict__ Wt,
                          int K, int N) {
  const int idx = blockIdx.x * blockDim.x + threadIdx.x;
  if (idx >= K * N) return;
  const int k = idx / N, n = idx % N;
  Wt[(long long)n * K + k] = (bfx)W[idx];
}

__global__ void cvt_bf16_k(const float* __restrict__ s, bfx* __restrict__ d,
                           long long n4) {
  const long long i = blockIdx.x * (long long)blockDim.x + threadIdx.x;
  if (i >= n4) return;
  const float4 f = reinterpret_cast<const float4*>(s)[i];
  v4bf o = {(bfx)f.x, (bfx)f.y, (bfx)f.z, (bfx)f.w};
  reinterpret_cast<v4bf*>(d)[i] = o;
}

__global__ void zero_f32_k(float* __restrict__ p, long long n) {
  const long long i = blockIdx.x * (long long)blockDim.x + threadIdx.x;
  if (i < n) p[i] = 0.0f;
}

// logits[e,h] = (q[dst]·(k[src]+eproj[e])) * 1/sqrt(16); running segment max
__global__ void logits_k(const float* __restrict__ q, const float* __restrict__ kk,
                         const float* __restrict__ ep,
                         const long long* __restrict__ srcI,
                         const long long* __restrict__ dstI,
                         float* __restrict__ logits, unsigned* __restrict__ mbuf,
                         int EH) {
  const int i = blockIdx.x * blockDim.x + threadIdx.x;
  if (i >= EH) return;
  const int ed = i >> 3, h = i & 7;
  const long long s = srcI[ed], d = dstI[ed];
  const float4* qp = reinterpret_cast<const float4*>(q + d * 128 + h * 16);
  const float4* kp = reinterpret_cast<const float4*>(kk + s * 128 + h * 16);
  const float4* epp =
      reinterpret_cast<const float4*>(ep + (long long)ed * 128 + h * 16);
  float acc = 0.f;
#pragma unroll
  for (int j = 0; j < 4; ++j) {
    const float4 a = qp[j], b = kp[j], c = epp[j];
    acc += a.x * (b.x + c.x) + a.y * (b.y + c.y) + a.z * (b.z + c.z) +
           a.w * (b.w + c.w);
  }
  const float lg = acc * 0.25f;
  logits[i] = lg;
  atomicMax(&mbuf[d * 8 + h], fkey(lg));
}

__global__ void expsum_k(float* __restrict__ logits,
                         const unsigned* __restrict__ mbuf,
                         float* __restrict__ sbuf,
                         const long long* __restrict__ dstI, int EH) {
  const int i = blockIdx.x * blockDim.x + threadIdx.x;
  if (i >= EH) return;
  const int ed = i >> 3, h = i & 7;
  const long long d = dstI[ed];
  const float e = expf(logits[i] - funkey(mbuf[d * 8 + h]));
  logits[i] = e;  // reuse buffer for exp values
  atomicAdd(&sbuf[d * 8 + h], e);
}

__global__ void msg_k(const float* __restrict__ ex, const float* __restrict__ sbuf,
                      const float* __restrict__ v,
                      const long long* __restrict__ srcI,
                      const long long* __restrict__ dstI, float* __restrict__ agg,
                      int EH) {
  const int i = blockIdx.x * blockDim.x + threadIdx.x;
  if (i >= EH) return;
  const int ed = i >> 3, h = i & 7;
  const long long s = srcI[ed], d = dstI[ed];
  const float a = ex[i] / sbuf[d * 8 + h];
  const float4* vp = reinterpret_cast<const float4*>(v + s * 128 + h * 16);
  float* ap = agg + d * 128 + h * 16;
#pragma unroll
  for (int j = 0; j < 4; ++j) {
    const float4 vv = vp[j];
    atomicAdd(ap + j * 4 + 0, a * vv.x);
    atomicAdd(ap + j * 4 + 1, a * vv.y);
    atomicAdd(ap + j * 4 + 2, a * vv.z);
    atomicAdd(ap + j * 4 + 3, a * vv.w);
  }
}

// ---------------------------------------------------------------------------
extern "C" void kernel_launch(void* const* d_in, const int* in_sizes, int n_in,
                              void* d_out, int out_size, void* d_ws,
                              size_t ws_size, hipStream_t stream) {
  (void)in_sizes; (void)n_in; (void)out_size; (void)ws_size;
  const float* x   = (const float*)d_in[0];
  const float* ea  = (const float*)d_in[1];
  const float* qw  = (const float*)d_in[2];  const float* qb  = (const float*)d_in[3];
  const float* kw  = (const float*)d_in[4];  const float* kb  = (const float*)d_in[5];
  const float* vw  = (const float*)d_in[6];  const float* vb  = (const float*)d_in[7];
  const float* ew  = (const float*)d_in[8];  const float* ebi = (const float*)d_in[9];
  const float* ow  = (const float*)d_in[10]; const float* ob  = (const float*)d_in[11];
  const float* f1w = (const float*)d_in[12]; const float* f1b = (const float*)d_in[13];
  const float* f2w = (const float*)d_in[14]; const float* f2b = (const float*)d_in[15];
  const float* u1w = (const float*)d_in[16]; const float* u1b = (const float*)d_in[17];
  const float* u2w = (const float*)d_in[18]; const float* u2b = (const float*)d_in[19];
  const float* ln1g = (const float*)d_in[20]; const float* ln1b = (const float*)d_in[21];
  const float* ln2g = (const float*)d_in[22]; const float* ln2b = (const float*)d_in[23];
  const float* lneg = (const float*)d_in[24]; const float* lneb = (const float*)d_in[25];
  const long long* ei = (const long long*)d_in[26];
  const long long* srcI = ei;
  const long long* dstI = ei + NEDGES;

  // --- workspace bump allocator (256B aligned) ---
  char* wp = (char*)d_ws;
  auto alloc = [&](size_t bytes) -> char* {
    char* p = wp;
    wp += (bytes + 255) & ~(size_t)255;
    return p;
  };
  bfx* xb   = (bfx*)alloc((size_t)NNODES * DIM * 2);
  bfx* eab  = (bfx*)alloc((size_t)NEDGES * DIM * 2);
  bfx* wtq  = (bfx*)alloc(128 * 128 * 2);
  bfx* wtk  = (bfx*)alloc(128 * 128 * 2);
  bfx* wtv  = (bfx*)alloc(128 * 128 * 2);
  bfx* wte  = (bfx*)alloc(128 * 128 * 2);
  bfx* wto  = (bfx*)alloc(128 * 128 * 2);
  bfx* wtf1 = (bfx*)alloc(256 * 128 * 2);
  bfx* wtf2 = (bfx*)alloc(128 * 256 * 2);
  bfx* wtu1 = (bfx*)alloc(128 * 384 * 2);
  bfx* wtu2 = (bfx*)alloc(128 * 128 * 2);
  float* qf  = (float*)alloc((size_t)NNODES * DIM * 4);
  float* kf  = (float*)alloc((size_t)NNODES * DIM * 4);
  float* vf  = (float*)alloc((size_t)NNODES * DIM * 4);
  float* epf = (float*)alloc((size_t)NEDGES * DIM * 4);
  float* lg  = (float*)alloc((size_t)NEDGES * 8 * 4);
  unsigned* mb = (unsigned*)alloc((size_t)NNODES * 8 * 4);
  float* sb  = (float*)alloc((size_t)NNODES * 8 * 4);
  float* agg = (float*)alloc((size_t)NNODES * DIM * 4);
  bfx* aggb  = (bfx*)alloc((size_t)NNODES * DIM * 2);
  float* hf  = (float*)alloc((size_t)NNODES * DIM * 4);
  bfx* hb    = (bfx*)alloc((size_t)NNODES * DIM * 2);
  bfx* mid   = (bfx*)alloc((size_t)NNODES * 256 * 2);
  bfx* h2b   = (bfx*)alloc((size_t)NNODES * DIM * 2);
  bfx* eum   = (bfx*)alloc((size_t)NEDGES * DIM * 2);

  float* h2out = (float*)d_out;                          // [N,128]
  float* eout  = (float*)d_out + (size_t)NNODES * DIM;   // [E,128]

  // --- weight prep (bf16 transpose) ---
  auto wt = [&](const float* W, bfx* Wt, int K, int N) {
    const int n = K * N;
    wt_bf16_k<<<(n + 255) / 256, 256, 0, stream>>>(W, Wt, K, N);
  };
  wt(qw, wtq, 128, 128);  wt(kw, wtk, 128, 128);  wt(vw, wtv, 128, 128);
  wt(ew, wte, 128, 128);  wt(ow, wto, 128, 128);
  wt(f1w, wtf1, 128, 256); wt(f2w, wtf2, 256, 128);
  wt(u1w, wtu1, 384, 128); wt(u2w, wtu2, 128, 128);

  // --- activation bf16 copies ---
  const long long nx4 = (long long)NNODES * DIM / 4;
  const long long ne4 = (long long)NEDGES * DIM / 4;
  cvt_bf16_k<<<(unsigned)((nx4 + 255) / 256), 256, 0, stream>>>(x, xb, nx4);
  cvt_bf16_k<<<(unsigned)((ne4 + 255) / 256), 256, 0, stream>>>(ea, eab, ne4);

  const dim3 gN((NNODES + 127) / 128, 1), gN2((NNODES + 127) / 128, 2);
  const dim3 gE((NEDGES + 127) / 128, 1);

  // --- q/k/v and edge projections (WMMA) ---
  gemm_wmma_k<0, false, false><<<gN, 128, 0, stream>>>(
      xb, 128, nullptr, nullptr, nullptr, wtq, qb, nullptr, nullptr, nullptr,
      qf, nullptr, NNODES, 128, 128);
  gemm_wmma_k<0, false, false><<<gN, 128, 0, stream>>>(
      xb, 128, nullptr, nullptr, nullptr, wtk, kb, nullptr, nullptr, nullptr,
      kf, nullptr, NNODES, 128, 128);
  gemm_wmma_k<0, false, false><<<gN, 128, 0, stream>>>(
      xb, 128, nullptr, nullptr, nullptr, wtv, vb, nullptr, nullptr, nullptr,
      vf, nullptr, NNODES, 128, 128);
  gemm_wmma_k<0, false, false><<<gE, 128, 0, stream>>>(
      eab, 128, nullptr, nullptr, nullptr, wte, ebi, nullptr, nullptr, nullptr,
      epf, nullptr, NEDGES, 128, 128);

  // --- scatter softmax attention ---
  const long long nmh = (long long)NNODES * 8;
  zero_f32_k<<<(unsigned)((nmh + 255) / 256), 256, 0, stream>>>((float*)mb, nmh);
  zero_f32_k<<<(unsigned)((nmh + 255) / 256), 256, 0, stream>>>(sb, nmh);
  const long long nag = (long long)NNODES * DIM;
  zero_f32_k<<<(unsigned)((nag + 255) / 256), 256, 0, stream>>>(agg, nag);

  const int EH = NEDGES * 8;
  logits_k<<<(EH + 255) / 256, 256, 0, stream>>>(qf, kf, epf, srcI, dstI, lg, mb, EH);
  expsum_k<<<(EH + 255) / 256, 256, 0, stream>>>(lg, mb, sb, dstI, EH);
  msg_k<<<(EH + 255) / 256, 256, 0, stream>>>(lg, sb, vf, srcI, dstI, agg, EH);
  cvt_bf16_k<<<(unsigned)((nx4 + 255) / 256), 256, 0, stream>>>(agg, aggb, nx4);

  // --- out-proj + (+x, LN1) -> h ---
  gemm_wmma_k<1, false, false><<<gN, 128, 0, stream>>>(
      aggb, 128, nullptr, nullptr, nullptr, wto, ob, x, ln1g, ln1b,
      hf, hb, NNODES, 128, 128);

  // --- FFN: gelu(h W1) W2 + (+h, LN2) -> h2 (output 1) ---
  gemm_wmma_k<0, false, true><<<gN2, 128, 0, stream>>>(
      hb, 128, nullptr, nullptr, nullptr, wtf1, f1b, nullptr, nullptr, nullptr,
      nullptr, mid, NNODES, 128, 256);
  gemm_wmma_k<1, false, false><<<gN, 128, 0, stream>>>(
      mid, 256, nullptr, nullptr, nullptr, wtf2, f2b, hf, ln2g, ln2b,
      h2out, h2b, NNODES, 256, 128);

  // --- edge update: gelu([h2[src]|h2[dst]|ea] U1) U2 + (+ea, LN_e) -> e_out ---
  gemm_wmma_k<0, true, true><<<gE, 128, 0, stream>>>(
      h2b, 128, eab, srcI, dstI, wtu1, u1b, nullptr, nullptr, nullptr,
      nullptr, eum, NEDGES, 384, 128);
  gemm_wmma_k<1, false, false><<<gE, 128, 0, stream>>>(
      eum, 128, nullptr, nullptr, nullptr, wtu2, u2b, ea, lneg, lneb,
      eout, nullptr, NEDGES, 128, 128);
}
